// GCLSTM_64458869178655
// MI455X (gfx1250) — matrix operand order repaired and checked by reading, not compiled, not run on hardware
//
#include <hip/hip_runtime.h>

// ---------------------------------------------------------------------------
// GC-LSTM cell for MI455X (gfx1250, wave32, WMMA).
//   N=50000 nodes, E=800000 edges, IN=32, OUT=256, K=4, lambda_max=2.0
// Dense part: M=N, K=1056, Ncols=1024 GEMM via v_wmma_f32_16x16x32_f16.
// A matrix pre-converted to f16 (read once per column quarter), staged to LDS
// with double-buffered global_load_async_to_lds_b128 (ASYNCcnt). Each wave
// owns all 4 gates for its tiles -> LSTM gating fused in registers.
// ---------------------------------------------------------------------------

typedef _Float16 h4  __attribute__((ext_vector_type(4)));
typedef _Float16 h8  __attribute__((ext_vector_type(8)));
typedef _Float16 h16 __attribute__((ext_vector_type(16)));
typedef float    f8  __attribute__((ext_vector_type(8)));

#define OUTC   256
#define INC    32
#define KTOT   1056          // 32 + 4*256
#define KCH    33            // 33 chunks of 32
#define MTILE  64

// ------------------------- sparse pipeline ---------------------------------

__global__ __launch_bounds__(256) void deg_kernel(const int* __restrict__ ei,
                                                  const float* __restrict__ ew,
                                                  float* __restrict__ deg, int E) {
    int i = blockIdx.x * 256 + threadIdx.x;
    if (i < E) unsafeAtomicAdd(&deg[ei[i]], ew[i]);
}

__global__ __launch_bounds__(256) void dis_kernel(float* __restrict__ deg, int n) {
    int i = blockIdx.x * 256 + threadIdx.x;
    if (i < n) { float d = deg[i]; deg[i] = (d > 0.f) ? __frsqrt_rn(d) : 0.f; }
}

// w_e = -(2/lam) * dis[src] * ew * dis[dst], lam = 2  ->  -dis*ew*dis
__global__ __launch_bounds__(256) void wgt_kernel(const int* __restrict__ ei,
                                                  const float* __restrict__ ew,
                                                  const float* __restrict__ dis,
                                                  float* __restrict__ w, int E) {
    int i = blockIdx.x * 256 + threadIdx.x;
    if (i < E) w[i] = -dis[ei[i]] * ew[i] * dis[ei[E + i]];
}

// out = -in (vectorized); seeds Tx2 = -Tx0, Tx3 = -Tx1 before scatter
__global__ __launch_bounds__(256) void negcopy_kernel(const float4* __restrict__ in,
                                                      float4* __restrict__ out, int n4) {
    int i = blockIdx.x * 256 + threadIdx.x;
    if (i < n4) { float4 v = in[i]; out[i] = make_float4(-v.x, -v.y, -v.z, -v.w); }
}

// out[dst] += scale * w[e] * x[src]; one wave (32 lanes) per edge, 8 ch/lane
__global__ __launch_bounds__(256) void scatter_kernel(const float* __restrict__ x,
                                                      const int* __restrict__ ei,
                                                      const float* __restrict__ w,
                                                      float scale,
                                                      float* __restrict__ out, int E) {
    int wave = threadIdx.x >> 5;
    int lane = threadIdx.x & 31;
    int e = blockIdx.x * 8 + wave;
    if (e >= E) return;
    int s = ei[e], d = ei[E + e];
    float c = scale * w[e];
    const float* xr = x + (size_t)s * OUTC;
    float* orow     = out + (size_t)d * OUTC;
#pragma unroll
    for (int i = 0; i < 8; ++i) {
        int ch = lane + 32 * i;
        unsafeAtomicAdd(&orow[ch], c * xr[ch]);
    }
}

// ------------------------- A-matrix f16 pre-pack ---------------------------
// Apack[row][k] f16, k in [0,1056) = [X(32) | H(256) | Tx1 | Tx2 | Tx3]
__global__ __launch_bounds__(256)
void aconv_kernel(const float* __restrict__ X, const float* __restrict__ H,
                  const float* __restrict__ Tx1, const float* __restrict__ Tx2,
                  const float* __restrict__ Tx3,
                  _Float16* __restrict__ Apack, int n) {
    int gidx = blockIdx.x * 256 + threadIdx.x;     // groups of 4 elements
    int total = n * (KTOT / 4);
    if (gidx >= total) return;
    int row = gidx / (KTOT / 4);
    int k = (gidx - row * (KTOT / 4)) * 4;
    const float* src;
    if (k < INC) src = X + (size_t)row * INC + k;
    else {
        int b = (k - INC) >> 8, off = (k - INC) & 255;
        const float* base = (b == 0) ? H : (b == 1) ? Tx1 : (b == 2) ? Tx2 : Tx3;
        src = base + (size_t)row * OUTC + off;
    }
    float4 v = *(const float4*)src;
    h4 o; o[0] = (_Float16)v.x; o[1] = (_Float16)v.y;
    o[2] = (_Float16)v.z; o[3] = (_Float16)v.w;
    *(h4*)(Apack + (size_t)row * KTOT + k) = o;
}

// ------------------------- weight packing ----------------------------------
// Per-lane B layout used here (contiguous K, as A) for v_wmma_f32_16x16x32_f16 (B is 32Kx16N):
//   lanes 0-15 : col = lane,    K = 0..15  (elems j=0..15)
//   lanes 16-31: col = lane-16, K = 16..31
// Buffer: Bpack[kc(33)][nt(64 = gate*16 + colsub)][lane(32)][j(16)]  f16
__global__ __launch_bounds__(256) void pack_kernel(const float* __restrict__ Wi, const float* __restrict__ Wf,
                                                   const float* __restrict__ Wc, const float* __restrict__ Wo,
                                                   const float* __restrict__ Ci, const float* __restrict__ Cf,
                                                   const float* __restrict__ Cc, const float* __restrict__ Co,
                                                   _Float16* __restrict__ Bpack) {
    int tid = blockIdx.x * 256 + threadIdx.x;       // = ((kc*64 + nt)*32 + lane)
    if (tid >= KCH * 64 * 32) return;
    int lane = tid & 31;
    int nt   = (tid >> 5) & 63;
    int kc   = tid >> 11;
    int g    = nt >> 4;
    int ncol = (nt & 15) * 16 + (lane & 15);
    int hi   = lane >> 4;
    const float* W  = (g == 0) ? Wi : (g == 1) ? Wf : (g == 2) ? Wc : Wo;
    const float* CW = (g == 0) ? Ci : (g == 1) ? Cf : (g == 2) ? Cc : Co;
#pragma unroll
    for (int j = 0; j < 16; ++j) {
        int k = kc * 32 + hi * 16 + j;              // global K row in [0,1056)
        float v = (k < INC) ? W[(size_t)k * OUTC + ncol]
                            : CW[(size_t)(k - INC) * OUTC + ncol];
        Bpack[(size_t)tid * 16 + j] = (_Float16)v;
    }
}

// ------------------------- fused WMMA GEMM + gating ------------------------
// Block: 256 threads (8 waves), 64 rows x 64 cols-per-gate x 4 gates.
// grid.x = ceil(N/64), grid.y = 4 (column quarter of each gate).
// Wave w: M-subtile ms = w>>1, col-subtiles {2*(w&1), 2*(w&1)+1}, ALL 4
// gates -> 8 accum tiles/wave, gating combined in registers.
__global__ __launch_bounds__(256)
void gemm_gate_kernel(const _Float16* __restrict__ Apack,
                      const _Float16* __restrict__ Bpack,
                      const float* __restrict__ Cin,
                      const float* __restrict__ bI, const float* __restrict__ bF,
                      const float* __restrict__ bC, const float* __restrict__ bO,
                      const float* __restrict__ cbI, const float* __restrict__ cbF,
                      const float* __restrict__ cbC, const float* __restrict__ cbO,
                      float* __restrict__ Hout, float* __restrict__ Cout, int n) {
    __shared__ __align__(16) _Float16 As[2][MTILE * 32];   // 8 KB double buffer

    const int t     = threadIdx.x;
    const int lane  = t & 31;
    const int wv    = t >> 5;
    const int hi    = lane >> 4;             // 0 or 1
    const int nlane = lane & 15;
    const int m0    = blockIdx.x * MTILE;
    const int cb    = blockIdx.y;            // 0..3 -> gate cols [cb*64, +64)

    const int ms = wv >> 1;                  // M-subtile of this wave
    const int s0 = (wv & 1) * 2, s1 = s0 + 1;
    const int ct0 = cb * 4 + s0, ct1 = cb * 4 + s1;

    f8 acc[2][4] = {};                       // [col pair][gate]

    // async A staging: thread t copies 16B: row rr, 16B-chunk q of the chunk
    const int rr = t >> 2, q = t & 3;
    int grow = m0 + rr; if (grow >= n) grow = n - 1;   // clamp (rows unused)
    const char* gRow = (const char*)Apack + (size_t)grow * (KTOT * 2) + q * 16;

    {   // prologue: chunk 0 -> buffer 0
        unsigned ldsoff = (unsigned)(size_t)&As[0][rr * 32 + q * 8];
        unsigned long long ga = (unsigned long long)gRow;
        asm volatile("global_load_async_to_lds_b128 %0, %1, off"
                     :: "v"(ldsoff), "v"(ga) : "memory");
    }

    for (int kc = 0; kc < KCH; ++kc) {
        const int buf = kc & 1;
        if (kc + 1 < KCH) {                  // prefetch next chunk
            unsigned ldsoff = (unsigned)(size_t)&As[buf ^ 1][rr * 32 + q * 8];
            unsigned long long ga = (unsigned long long)(gRow + (size_t)(kc + 1) * 64);
            asm volatile("global_load_async_to_lds_b128 %0, %1, off"
                         :: "v"(ldsoff), "v"(ga) : "memory");
            asm volatile("s_wait_asynccnt 0x1" ::: "memory");
        } else {
            asm volatile("s_wait_asynccnt 0x0" ::: "memory");
        }
        __syncthreads();                     // all waves' copies visible

        // A operand (16x32 f16): lane half 'hi' holds K{hi*16 .. hi*16+15}
        const int m = ms * 16 + nlane;
        h8 alo = *(const h8*)&As[buf][m * 32 + hi * 16];
        h8 ahi = *(const h8*)&As[buf][m * 32 + hi * 16 + 8];
        h16 a = __builtin_shufflevector(alo, ahi, 0, 1, 2, 3, 4, 5, 6, 7,
                                        8, 9, 10, 11, 12, 13, 14, 15);

        const _Float16* bbase = Bpack + (size_t)kc * 64 * 512;
#pragma unroll
        for (int g = 0; g < 4; ++g) {
            h16 b0 = *(const h16*)(bbase + ((size_t)(g * 16 + ct0) * 32 + lane) * 16);
            h16 b1 = *(const h16*)(bbase + ((size_t)(g * 16 + ct1) * 32 + lane) * 16);
            acc[0][g] = __builtin_amdgcn_wmma_f32_16x16x32_f16(
                false, a, false, b0, (short)0, acc[0][g], false, false);
            acc[1][g] = __builtin_amdgcn_wmma_f32_16x16x32_f16(
                false, a, false, b1, (short)0, acc[1][g], false, false);
        }
        __syncthreads();                     // done reading buf before rewrite
    }

    // --- in-register LSTM gating epilogue (D layout: elem r -> M=r+8*hi) ---
    const int c0 = cb * 64;
#pragma unroll
    for (int p = 0; p < 2; ++p) {
        const int s = (p == 0) ? s0 : s1;
        const int gc = c0 + s * 16 + nlane;
        const float bi_ = bI[gc] + cbI[gc];
        const float bf_ = bF[gc] + cbF[gc];
        const float bc_ = bC[gc] + cbC[gc];
        const float bo_ = bO[gc] + cbO[gc];
#pragma unroll
        for (int r = 0; r < 8; ++r) {
            int gr = m0 + ms * 16 + hi * 8 + r;
            if (gr >= n) continue;
            size_t off = (size_t)gr * OUTC + gc;
            float gi = acc[p][0][r] + bi_;
            float gf = acc[p][1][r] + bf_;
            float gg = acc[p][2][r] + bc_;
            float go = acc[p][3][r] + bo_;
            float I = 1.f / (1.f + __expf(-gi));
            float F = 1.f / (1.f + __expf(-gf));
            float T = tanhf(gg);
            float O = 1.f / (1.f + __expf(-go));
            float c = F * Cin[off] + I * T;
            Hout[off] = O * tanhf(c);
            Cout[off] = c;
        }
    }
}

// ------------------------- launch ------------------------------------------

extern "C" void kernel_launch(void* const* d_in, const int* in_sizes, int n_in,
                              void* d_out, int out_size, void* d_ws, size_t ws_size,
                              hipStream_t stream) {
    const float* X  = (const float*)d_in[0];
    const int*   ei = (const int*)d_in[1];
    const float* ew = (const float*)d_in[2];
    const float* H  = (const float*)d_in[3];
    const float* C  = (const float*)d_in[4];
    const float* Wi = (const float*)d_in[5],  *bi = (const float*)d_in[6];
    const float* cwi = (const float*)d_in[7], *cbi = (const float*)d_in[8];
    const float* Wf = (const float*)d_in[9],  *bf = (const float*)d_in[10];
    const float* cwf = (const float*)d_in[11], *cbf = (const float*)d_in[12];
    const float* Wc = (const float*)d_in[13], *bc = (const float*)d_in[14];
    const float* cwc = (const float*)d_in[15], *cbc = (const float*)d_in[16];
    const float* Wo = (const float*)d_in[17], *bo = (const float*)d_in[18];
    const float* cwo = (const float*)d_in[19], *cbo = (const float*)d_in[20];

    const int E = in_sizes[2];
    const int N = in_sizes[3] / OUTC;
    const size_t feat = (size_t)N * OUTC;

    // workspace carve-up (bytes), 256B-aligned slots
    char* ws = (char*)d_ws;
    size_t o = 0;
    float* deg = (float*)(ws + o); o += (((size_t)N * 4) + 255) & ~(size_t)255;
    float* w   = (float*)(ws + o); o += (((size_t)E * 4) + 255) & ~(size_t)255;
    float* Tx1 = (float*)(ws + o); o += (feat * 4 + 255) & ~(size_t)255;
    float* Tx2 = (float*)(ws + o); o += (feat * 4 + 255) & ~(size_t)255;
    float* Tx3 = (float*)(ws + o); o += (feat * 4 + 255) & ~(size_t)255;
    _Float16* Apack = (_Float16*)(ws + o); o += ((size_t)N * KTOT * 2 + 255) & ~(size_t)255;
    _Float16* Bpack = (_Float16*)(ws + o);

    // 1) degree -> inverse sqrt -> scaled edge weights
    hipMemsetAsync(deg, 0, (size_t)N * 4, stream);
    deg_kernel<<<(E + 255) / 256, 256, 0, stream>>>(ei, ew, deg, E);
    dis_kernel<<<(N + 255) / 256, 256, 0, stream>>>(deg, N);
    wgt_kernel<<<(E + 255) / 256, 256, 0, stream>>>(ei, ew, deg, w, E);

    // 2) Chebyshev recurrence (diagonal term is 0 for lambda_max = 2):
    //    Tx1 = prop(H); Tx2 = 2*prop(Tx1) - H; Tx3 = 2*prop(Tx2) - Tx1
    const int n4 = (int)(feat / 4);
    hipMemsetAsync(Tx1, 0, feat * 4, stream);
    scatter_kernel<<<(E + 7) / 8, 256, 0, stream>>>(H, ei, w, 1.f, Tx1, E);
    negcopy_kernel<<<(n4 + 255) / 256, 256, 0, stream>>>((const float4*)H, (float4*)Tx2, n4);
    scatter_kernel<<<(E + 7) / 8, 256, 0, stream>>>(Tx1, ei, w, 2.f, Tx2, E);
    negcopy_kernel<<<(n4 + 255) / 256, 256, 0, stream>>>((const float4*)Tx1, (float4*)Tx3, n4);
    scatter_kernel<<<(E + 7) / 8, 256, 0, stream>>>(Tx2, ei, w, 2.f, Tx3, E);

    // 3) pre-convert A to f16 (read once) + pack weights to WMMA B layout
    {
        int total = N * (KTOT / 4);
        aconv_kernel<<<(total + 255) / 256, 256, 0, stream>>>(X, H, Tx1, Tx2, Tx3,
                                                              Apack, N);
    }
    pack_kernel<<<(KCH * 64 * 32 + 255) / 256, 256, 0, stream>>>(
        Wi, Wf, Wc, Wo, cwi, cwf, cwc, cwo, Bpack);

    // 4) fused WMMA GEMM (all 4 gates) + LSTM gating
    float* Hout = (float*)d_out;
    float* Cout = (float*)d_out + feat;
    dim3 grid((N + MTILE - 1) / MTILE, 4);
    gemm_gate_kernel<<<grid, 256, 0, stream>>>(Apack, Bpack, C,
                                               bi, bf, bc, bo, cbi, cbf, cbc, cbo,
                                               Hout, Cout, N);
}
